// STA_9311489098021
// MI455X (gfx1250) — compile-verified, matrix operand order checked by reading, not run on hardware
//
#include <hip/hip_runtime.h>
#include <hip/hip_bf16.h>
#include <math.h>

typedef __attribute__((ext_vector_type(16))) _Float16 v16h;
typedef __attribute__((ext_vector_type(8)))  _Float16 v8h;
typedef __attribute__((ext_vector_type(8)))  float    v8f;
typedef __attribute__((ext_vector_type(4)))  int      v4i;

// ---------------------------------------------------------------------------
// CDNA5 async global->LDS copy (ASYNCcnt path), guarded so compile never breaks
// ---------------------------------------------------------------------------
#if defined(__HIP_DEVICE_COMPILE__) && __has_builtin(__builtin_amdgcn_global_load_async_to_lds_b128)
#define USE_ASYNC_LDS 1
#else
#define USE_ASYNC_LDS 0
#endif

__device__ __forceinline__ void async_copy_b128(const void* g, void* l) {
#if USE_ASYNC_LDS
    // builtin signature (from hipcc diagnostic): (int4 AS1*, int4 AS3*, imm, cpol)
    __builtin_amdgcn_global_load_async_to_lds_b128(
        (__attribute__((address_space(1))) v4i*)g,
        (__attribute__((address_space(3))) v4i*)l, 0, 0);
#else
    *(v8h*)l = *(const v8h*)g;  // fallback: b128 load + b128 ds_store
#endif
}

__device__ __forceinline__ void wait_async_lds() {
#if USE_ASYNC_LDS
#if __has_builtin(__builtin_amdgcn_s_wait_asynccnt)
    __builtin_amdgcn_s_wait_asynccnt(0);
#else
    asm volatile("s_wait_asynccnt 0x0" ::: "memory");
#endif
#endif
}

// ---------------------------------------------------------------------------
// WMMA helpers (gfx1250 wave32, v_wmma_f32_16x16x32_f16)
// ---------------------------------------------------------------------------
__device__ __forceinline__ v8f wmma_f16(v16h a, v16h b, v8f c) {
    return __builtin_amdgcn_wmma_f32_16x16x32_f16(false, a, false, b, (short)0, c,
                                                  false, false);
}

// ISA 7.12.2: lane l(0..15)=row M=l holds K={0..7,16..23}; lanes 16..31 hold the
// same rows with K={8..15,24..31}.  With a k-major LDS tile (pitch in halves,
// multiple of 8) each lane's fragment is TWO contiguous 16B runs -> 2x ds_load_b128.
// Used for A (row=M) and for B staged TRANSPOSED in LDS (row=N).
__device__ __forceinline__ v16h load_frag(const _Float16* lds, int pitch,
                                          int rowOff, int lane) {
    const int row = rowOff + (lane & 15);
    const int kl  = (lane & 16) ? 8 : 0;
    const _Float16* p = lds + row * pitch + kl;
    v8h lo = *(const v8h*)(p);
    v8h hi = *(const v8h*)(p + 16);
    return __builtin_shufflevector(lo, hi, 0, 1, 2, 3, 4, 5, 6, 7,
                                   8, 9, 10, 11, 12, 13, 14, 15);
}

#define TPITCH 40  // halves; multiple of 8 keeps 16B alignment, breaks bank conflicts

// ---------------------------------------------------------------------------
// Batched f16 GEMM: C[M,N] = A[M,K] * B[K,N]  (f32 or f16 out)
// TA=true: logical A'[m,k] = A[k*lda + m]   (gram  Xe^T Xe)
// 128 threads (4 waves), 64x64 tile, K step 32, double-buffered LDS with
// async A-tile copies overlapped against WMMA on the other buffer.
// All M,N multiples of 64; K, lda, ldb multiples of 32/8 (true for every call).
// ---------------------------------------------------------------------------
template <bool TA, bool OUT16>
__global__ __launch_bounds__(128) void gemm_wmma(
    const _Float16* __restrict__ A, const _Float16* __restrict__ B,
    void* __restrict__ Cv, int Kd, int lda, int ldb, int ldc,
    long long sA, long long sB, long long sC) {
    __shared__ alignas(16) _Float16 As[2][64 * TPITCH];
    __shared__ alignas(16) _Float16 Bs[2][64 * TPITCH];
    const int tid  = threadIdx.x;
    const int lane = tid & 31;
    const int wid  = tid >> 5;
    const int m0   = blockIdx.y * 64;
    const int n0   = blockIdx.x * 64;
    const long long bz = blockIdx.z;
    A += bz * sA;
    B += bz * sB;
    const int wm = (wid >> 1) * 32;
    const int wn = (wid & 1) * 32;

    // A tile: 64(m) x 32(k), k-major.
    auto stageA = [&](int buf, int kt) {
        if (TA) {
            // coalesced over m: vector global load, strided LDS scatter
#pragma unroll
            for (int i = 0; i < 2; ++i) {
                int q = tid + i * 128;        // 256 chunks of 8 halves
                int c = q >> 3, r8 = q & 7;   // k-col, m-octet
                v8h g = *(const v8h*)&A[(long long)(kt + c) * lda + m0 + r8 * 8];
#pragma unroll
                for (int j = 0; j < 8; ++j)
                    As[buf][(r8 * 8 + j) * TPITCH + c] = g[j];
            }
        } else {
            // rows are contiguous in memory -> async b128 straight into LDS
#pragma unroll
            for (int i = 0; i < 2; ++i) {
                int q = tid + i * 128;
                int r = q >> 2, c8 = q & 3;   // m-row, 8-half chunk in k
                async_copy_b128(&A[(long long)(m0 + r) * lda + kt + c8 * 8],
                                &As[buf][r * TPITCH + c8 * 8]);
            }
        }
    };
    // B tile staged TRANSPOSED: Bs[n][k]; coalesced global reads over n.
    auto stageB = [&](int buf, int kt) {
#pragma unroll
        for (int i = 0; i < 2; ++i) {
            int q = tid + i * 128;
            int k = q >> 3, c8 = q & 7;       // k-row, n-octet
            v8h g = *(const v8h*)&B[(long long)(kt + k) * ldb + n0 + c8 * 8];
#pragma unroll
            for (int j = 0; j < 8; ++j)
                Bs[buf][(c8 * 8 + j) * TPITCH + k] = g[j];
        }
    };

    v8f acc00 = {0.f,0.f,0.f,0.f,0.f,0.f,0.f,0.f};
    v8f acc01 = acc00, acc10 = acc00, acc11 = acc00;

    stageB(0, 0);
    stageA(0, 0);
    wait_async_lds();
    __syncthreads();

    int cur = 0;
    for (int kt = 0; kt < Kd; kt += 32) {
        const int nxt = cur ^ 1;
        if (kt + 32 < Kd) {  // prefetch next tile while WMMAs run on `cur`
            stageB(nxt, kt + 32);
            stageA(nxt, kt + 32);
        }
        v16h a0 = load_frag(As[cur], TPITCH, wm, lane);
        v16h a1 = load_frag(As[cur], TPITCH, wm + 16, lane);
        v16h b0 = load_frag(Bs[cur], TPITCH, wn, lane);
        v16h b1 = load_frag(Bs[cur], TPITCH, wn + 16, lane);
        acc00 = wmma_f16(a0, b0, acc00);
        acc01 = wmma_f16(a0, b1, acc01);
        acc10 = wmma_f16(a1, b0, acc10);
        acc11 = wmma_f16(a1, b1, acc11);
        wait_async_lds();
        __syncthreads();
        cur = nxt;
    }

    v8f accs[4] = {acc00, acc01, acc10, acc11};
#pragma unroll
    for (int s = 0; s < 4; ++s) {
        const int sm = (s >> 1) * 16, sn = (s & 1) * 16;
#pragma unroll
        for (int r = 0; r < 8; ++r) {
            int m = m0 + wm + sm + r + ((lane & 16) ? 8 : 0);
            int n = n0 + wn + sn + (lane & 15);
            float v = accs[s][r];
            if (OUT16)
                ((_Float16*)Cv)[bz * sC + (long long)m * ldc + n] = (_Float16)v;
            else
                ((float*)Cv)[bz * sC + (long long)m * ldc + n] = v;
        }
    }
}

// ---------------------------------------------------------------------------
// Conv as WMMA GEMM. W==1 with 1-pixel padding => only kw==1 taps touch data:
//   Out[b,co,h] = bias[co] + sum_{kh,ci} Wf[co, kh*1024+ci] * Hat[b,ci,h+kh-1]
// M=1024 (co), N=512 (h), K=3072.  Weight tiles go through the async path;
// the shifted/zero-padded Hat reads are staged transposed, BRANCHLESS:
// clamp the address (always in-bounds) and cndmask the value to zero.
// ---------------------------------------------------------------------------
__global__ __launch_bounds__(128) void conv_wmma(
    const _Float16* __restrict__ Wf,   // [1024][3072]
    const _Float16* __restrict__ Hat,  // [16][1024][512]
    const float* __restrict__ bias,    // [1024]
    float* __restrict__ Out,           // pre-offset to channel base
    long long outBatchStride) {
    __shared__ alignas(16) _Float16 As[2][64 * TPITCH];
    __shared__ alignas(16) _Float16 Bs[2][64 * TPITCH];
    const int tid  = threadIdx.x;
    const int lane = tid & 31;
    const int wid  = tid >> 5;
    const int m0   = blockIdx.y * 64;  // co
    const int n0   = blockIdx.x * 64;  // h
    Hat += (long long)blockIdx.z * (1024LL * 512);
    Out += (long long)blockIdx.z * outBatchStride;
    const int wm = (wid >> 1) * 32;
    const int wn = (wid & 1) * 32;

    auto stageA = [&](int buf, int kt) {
#pragma unroll
        for (int i = 0; i < 2; ++i) {
            int q = tid + i * 128;
            int r = q >> 2, c8 = q & 3;
            async_copy_b128(&Wf[(long long)(m0 + r) * 3072 + kt + c8 * 8],
                            &As[buf][r * TPITCH + c8 * 8]);
        }
    };
    auto stageB = [&](int buf, int kt) {
#pragma unroll
        for (int i = 0; i < 16; ++i) {
            int idx = tid + i * 128;      // 2048 elements
            int k = idx >> 6, n = idx & 63;
            int kc = kt + k;
            int kh = kc >> 10;
            int ci = kc & 1023;
            int h  = n0 + n + kh - 1;            // shifted tap
            int hc = min(max(h, 0), 511);        // address-safe clamp
            _Float16 v = Hat[ci * 512 + hc];     // unconditional load
            Bs[buf][n * TPITCH + k] =
                ((unsigned)h < 512u) ? v : (_Float16)0.0f;  // v_cndmask, no branch
        }
    };

    v8f acc00 = {0.f,0.f,0.f,0.f,0.f,0.f,0.f,0.f};
    v8f acc01 = acc00, acc10 = acc00, acc11 = acc00;

    stageB(0, 0);
    stageA(0, 0);
    wait_async_lds();
    __syncthreads();

    int cur = 0;
    for (int kt = 0; kt < 3072; kt += 32) {
        const int nxt = cur ^ 1;
        if (kt + 32 < 3072) {
            stageB(nxt, kt + 32);
            stageA(nxt, kt + 32);
        }
        v16h a0 = load_frag(As[cur], TPITCH, wm, lane);
        v16h a1 = load_frag(As[cur], TPITCH, wm + 16, lane);
        v16h b0 = load_frag(Bs[cur], TPITCH, wn, lane);
        v16h b1 = load_frag(Bs[cur], TPITCH, wn + 16, lane);
        acc00 = wmma_f16(a0, b0, acc00);
        acc01 = wmma_f16(a0, b1, acc01);
        acc10 = wmma_f16(a1, b0, acc10);
        acc11 = wmma_f16(a1, b1, acc11);
        wait_async_lds();
        __syncthreads();
        cur = nxt;
    }

    v8f accs[4] = {acc00, acc01, acc10, acc11};
#pragma unroll
    for (int s = 0; s < 4; ++s) {
        const int sm = (s >> 1) * 16, sn = (s & 1) * 16;
#pragma unroll
        for (int r = 0; r < 8; ++r) {
            int m = m0 + wm + sm + r + ((lane & 16) ? 8 : 0);
            int n = n0 + wn + sn + (lane & 15);
            Out[(long long)m * 512 + n] = accs[s][r] + bias[m];
        }
    }
}

// ---------------------------------------------------------------------------
// Small helper kernels
// ---------------------------------------------------------------------------
__global__ void f32_to_f16(const float* __restrict__ in,
                           _Float16* __restrict__ out, long long n) {
    long long i = (long long)blockIdx.x * blockDim.x + threadIdx.x;
    long long stride = (long long)gridDim.x * blockDim.x;
    for (; i < n; i += stride) out[i] = (_Float16)in[i];
}

// wfold[co][kh][ci] = w[co][ci][kh][1]  (only kw==1 matters since W==1)
__global__ void fold_conv_w(const float* __restrict__ w,
                            _Float16* __restrict__ wf) {
    long long n = 1024LL * 3 * 1024;
    long long i = (long long)blockIdx.x * blockDim.x + threadIdx.x;
    long long stride = (long long)gridDim.x * blockDim.x;
    for (; i < n; i += stride) {
        int co = (int)(i / 3072);
        int rem = (int)(i % 3072);
        int kh = rem >> 10;
        int ci = rem & 1023;
        wf[i] = (_Float16)w[((long long)(co * 1024 + ci) * 3 + kh) * 3 + 1];
    }
}

// Row softmax over N=512, f32 in -> f16 out. One 128-thread block per row.
__global__ __launch_bounds__(128) void softmax_rows(
    const float* __restrict__ G, _Float16* __restrict__ A) {
    __shared__ float red[128];
    const int tid = threadIdx.x;
    const float* g = G + (long long)blockIdx.x * 512;
    _Float16* a = A + (long long)blockIdx.x * 512;

    float m = -3.4e38f;
    for (int j = tid; j < 512; j += 128) m = fmaxf(m, g[j]);
    red[tid] = m;
    __syncthreads();
    for (int off = 64; off > 0; off >>= 1) {
        if (tid < off) red[tid] = fmaxf(red[tid], red[tid + off]);
        __syncthreads();
    }
    const float mx = red[0];
    __syncthreads();

    float sum = 0.f;
    for (int j = tid; j < 512; j += 128) sum += __expf(g[j] - mx);
    red[tid] = sum;
    __syncthreads();
    for (int off = 64; off > 0; off >>= 1) {
        if (tid < off) red[tid] += red[tid + off];
        __syncthreads();
    }
    const float inv = 1.0f / red[0];
    for (int j = tid; j < 512; j += 128)
        a[j] = (_Float16)(__expf(g[j] - mx) * inv);
}

// Hat[b,c,k]: c<512 -> M[b,c,k]*V[b,c,k] ; c>=512 -> V[b,c-512,k]; +sin(k) opt
template <bool ADD_SIN>
__global__ void build_hat(const float* __restrict__ M,
                          const float* __restrict__ V,
                          _Float16* __restrict__ Hat) {
    const long long n = 16LL * 1024 * 512;
    long long i = (long long)blockIdx.x * blockDim.x + threadIdx.x;
    long long stride = (long long)gridDim.x * blockDim.x;
    for (; i < n; i += stride) {
        int b = (int)(i >> 19);
        int rem = (int)(i & ((1 << 19) - 1));
        int c = rem >> 9;
        int k = rem & 511;
        long long src = (long long)b * 512 * 512 + (long long)(c & 511) * 512 + k;
        float v = (c < 512) ? M[src] * V[src] : V[src];
        if (ADD_SIN) v += __sinf((float)k);
        Hat[i] = (_Float16)v;
    }
}

// ---------------------------------------------------------------------------
// Host launcher
// ---------------------------------------------------------------------------
extern "C" void kernel_launch(void* const* d_in, const int* in_sizes, int n_in,
                              void* d_out, int out_size, void* d_ws,
                              size_t ws_size, hipStream_t stream) {
    const float* X   = (const float*)d_in[0];
    const float* S   = (const float*)d_in[1];
    const float* W1x = (const float*)d_in[2];
    const float* W1s = (const float*)d_in[3];
    const float* W2x = (const float*)d_in[4];
    const float* W2s = (const float*)d_in[5];
    const float* cwx = (const float*)d_in[6];
    const float* cbx = (const float*)d_in[7];
    const float* cws = (const float*)d_in[8];
    const float* cbs = (const float*)d_in[9];
    float* out = (float*)d_out;

    char* cur = (char*)d_ws;
    auto alloc = [&](size_t bytes) -> void* {
        void* p = (void*)cur;
        cur += (bytes + 255) & ~(size_t)255;
        return p;
    };
    const long long EL = 16LL * 512 * 512;
    _Float16* Xh   = (_Float16*)alloc(EL * 2);
    _Float16* Sh   = (_Float16*)alloc(EL * 2);
    _Float16* W1xh = (_Float16*)alloc(512LL * 512 * 2);
    _Float16* W1sh = (_Float16*)alloc(512LL * 512 * 2);
    _Float16* W2xh = (_Float16*)alloc(512LL * 512 * 2);
    _Float16* W2sh = (_Float16*)alloc(512LL * 512 * 2);
    _Float16* Wfx  = (_Float16*)alloc(1024LL * 3072 * 2);
    _Float16* Wfs  = (_Float16*)alloc(1024LL * 3072 * 2);
    _Float16* Xeh  = (_Float16*)alloc(EL * 2);
    _Float16* Seh  = (_Float16*)alloc(EL * 2);
    _Float16* Ah   = (_Float16*)alloc(EL * 2);                // softmax (reused)
    _Float16* XAh  = (_Float16*)alloc(EL * 2);                // Xa/Sa (reused)
    float*    Gbuf = (float*)alloc(EL * 4);                   // gram then M (reused)
    _Float16* Hat  = (_Float16*)alloc(16LL * 1024 * 512 * 2); // reused x/s

    const long long sAct = 512LL * 512;

    f32_to_f16<<<2048, 256, 0, stream>>>(X, Xh, EL);
    f32_to_f16<<<2048, 256, 0, stream>>>(S, Sh, EL);
    f32_to_f16<<<512, 256, 0, stream>>>(W1x, W1xh, 512LL * 512);
    f32_to_f16<<<512, 256, 0, stream>>>(W1s, W1sh, 512LL * 512);
    f32_to_f16<<<512, 256, 0, stream>>>(W2x, W2xh, 512LL * 512);
    f32_to_f16<<<512, 256, 0, stream>>>(W2s, W2sh, 512LL * 512);
    fold_conv_w<<<3072, 256, 0, stream>>>(cwx, Wfx);
    fold_conv_w<<<3072, 256, 0, stream>>>(cws, Wfs);

    dim3 gemmGrid(8, 8, 16);
    dim3 convGrid(8, 16, 16);

    // cross-wired projections: Xe = S@W1x, Se = X@W1s
    gemm_wmma<false, true><<<gemmGrid, 128, 0, stream>>>(
        Sh, W1xh, Xeh, 512, 512, 512, 512, sAct, 0, sAct);
    gemm_wmma<false, true><<<gemmGrid, 128, 0, stream>>>(
        Xh, W1sh, Seh, 512, 512, 512, 512, sAct, 0, sAct);

    // ===== X path =====
    gemm_wmma<true, false><<<gemmGrid, 128, 0, stream>>>(
        Xeh, Xeh, Gbuf, 512, 512, 512, 512, sAct, sAct, sAct);     // G = Xe^T Xe
    softmax_rows<<<16 * 512, 128, 0, stream>>>(Gbuf, Ah);
    gemm_wmma<false, true><<<gemmGrid, 128, 0, stream>>>(
        Xeh, Ah, XAh, 512, 512, 512, 512, sAct, sAct, sAct);       // Xa = Xe@A
    gemm_wmma<false, false><<<gemmGrid, 128, 0, stream>>>(
        XAh, W2xh, Gbuf, 512, 512, 512, 512, sAct, 0, sAct);       // Mx
    build_hat<false><<<4096, 256, 0, stream>>>(Gbuf, X, Hat);
    conv_wmma<<<convGrid, 128, 0, stream>>>(Wfx, Hat, cbx, out, 2048LL * 512);

    // ===== S path =====
    gemm_wmma<true, false><<<gemmGrid, 128, 0, stream>>>(
        Seh, Seh, Gbuf, 512, 512, 512, 512, sAct, sAct, sAct);
    softmax_rows<<<16 * 512, 128, 0, stream>>>(Gbuf, Ah);
    gemm_wmma<false, true><<<gemmGrid, 128, 0, stream>>>(
        Seh, Ah, XAh, 512, 512, 512, 512, sAct, sAct, sAct);
    gemm_wmma<false, false><<<gemmGrid, 128, 0, stream>>>(
        XAh, W2sh, Gbuf, 512, 512, 512, 512, sAct, 0, sAct);
    build_hat<true><<<4096, 256, 0, stream>>>(Gbuf, S, Hat);       // + sin(k)
    conv_wmma<<<convGrid, 128, 0, stream>>>(Wfs, Hat, cbs,
                                            out + 1024LL * 512, 2048LL * 512);
}
